// GraNDAG_11467562680735
// MI455X (gfx1250) — compile-verified, matrix operand order checked by reading, not run on hardware
//
#include <hip/hip_runtime.h>

typedef __attribute__((ext_vector_type(16))) __bf16 v16bf;
typedef __attribute__((ext_vector_type(8)))  __bf16 v8bf;
typedef __attribute__((ext_vector_type(8)))  float  v8f;

#define Dn 128
#define Hn 512
#define Nn 2048
#define AP  520          // activation row pitch (bf16 elems): 1040 B rows
#define WTP 40           // transposed-weight row pitch (bf16 elems): 80 B rows
#define NTHREADS 256
#define CHUNK_ELEMS (Hn * WTP)          // 20480 bf16 = 40960 B per 32-K chunk block
#define CHUNK_BYTES (CHUNK_ELEMS * 2)
#define FT_N 64          // fast-path rows per workgroup
#define FB_N 32          // fallback rows per workgroup

static __device__ __forceinline__ __bf16 f2b(float f) { return (__bf16)f; }

// wave-relative LDS byte address: low 32 bits of the generic pointer (LDS aperture)
static __device__ __forceinline__ unsigned lds_addr(const void* p) {
  return (unsigned)(unsigned long long)p;
}

// Build a v16bf fragment: elems [0..7]=p[0..7], [8..15]=p[16..23] (both 16B-aligned)
static __device__ __forceinline__ v16bf load_frag(const __bf16* p) {
  v8bf lo = *(const v8bf*)(p);
  v8bf hi = *(const v8bf*)(p + 16);
  return __builtin_shufflevector(lo, hi, 0,1,2,3,4,5,6,7,8,9,10,11,12,13,14,15);
}

// Async-copy one contiguous 40,960 B chunk (pre-laid-out WT image) global -> LDS.
static __device__ __forceinline__ void copy_chunk_async(unsigned lds_base,
                                                        const __bf16* g, int tid) {
  const char* gp = (const char*)g + tid * 16;
  unsigned l = lds_base + tid * 16;
  #pragma unroll
  for (int i = 0; i < 10; ++i) {
    asm volatile("global_load_async_to_lds_b128 %0, %1, off"
                 :: "v"(l + i * 4096), "v"(gp + (long)i * 4096) : "memory");
  }
}

// ---------------- precompute: fp32 W -> bf16 LDS-image blocks in ws ----------------
__global__ __launch_bounds__(NTHREADS)
void convert_weights_kernel(const float* __restrict__ W0,
                            const float* __restrict__ W1,
                            __bf16* __restrict__ ws) {
  __shared__ __align__(16) __bf16 sWT[CHUNK_ELEMS];
  const int tid  = threadIdx.x;
  const int d    = blockIdx.y;
  const int cidx = blockIdx.x;          // 0..3 -> layer0 chunk, 4..19 -> layer1 chunk
  const float* Wg;
  int ch; long blk;
  if (cidx < 4) { Wg = W0 + (long)d * Dn * Hn; ch = cidx;     blk = (long)d * 4 + ch; }
  else          { Wg = W1 + (long)d * Hn * Hn; ch = cidx - 4; blk = (long)Dn * 4 + (long)d * 16 + ch; }

  #pragma unroll
  for (int i = 0; i < 16; ++i) {        // zero pad cols k=32..39
    int idx = tid + i * NTHREADS;
    sWT[(idx >> 3) * WTP + 32 + (idx & 7)] = f2b(0.f);
  }
  #pragma unroll
  for (int i = 0; i < 16; ++i) {        // coalesced fp32 read, transpose+convert
    int idx = tid + i * NTHREADS;
    int k   = idx >> 7;
    int n4  = (idx & 127) << 2;
    float4 v = *(const float4*)(Wg + (long)(ch * 32 + k) * Hn + n4);
    sWT[(n4 + 0) * WTP + k] = f2b(v.x);
    sWT[(n4 + 1) * WTP + k] = f2b(v.y);
    sWT[(n4 + 2) * WTP + k] = f2b(v.z);
    sWT[(n4 + 3) * WTP + k] = f2b(v.w);
  }
  __syncthreads();
  const float4* src = (const float4*)sWT;
  float4* dst = (float4*)ws + blk * (CHUNK_BYTES / 16);
  #pragma unroll
  for (int i = 0; i < 10; ++i) dst[tid + i * NTHREADS] = src[tid + i * NTHREADS];
}

// ---------------- fast path: one fused layer from pre-converted ws blocks ----------------
// sA[:, 0:K] (64 rows, bf16) @ W[d] -> leaky_relu -> sA[:, 0:512].
template<int K>
static __device__ __forceinline__ void lc_gemm_ws(const __bf16* __restrict__ wblk,
                                                  __bf16* sA, __bf16* sWT,
                                                  int tid, int wave, int lane) {
  constexpr int NCH = K / 32;
  v8f acc[4][4] = {};
  const int m  = lane & 15;
  const int hk = (lane >> 4) << 3;
  const unsigned wt0 = lds_addr(sWT);

  copy_chunk_async(wt0, wblk, tid);              // chunk 0 -> buf 0
  #pragma unroll 1
  for (int ch = 0; ch < NCH; ++ch) {
    const int cur = ch & 1;
    if (ch + 1 < NCH) {                          // overlap next copy with this chunk's WMMAs
      copy_chunk_async(wt0 + (1 - cur) * CHUNK_BYTES,
                       wblk + (long)(ch + 1) * CHUNK_ELEMS, tid);
      asm volatile("s_wait_asynccnt 10" ::: "memory");   // previous batch (10) retired
    } else {
      asm volatile("s_wait_asynccnt 0" ::: "memory");
    }
    __syncthreads();                             // buf[cur] visible to all waves
    const __bf16* WTc = sWT + cur * CHUNK_ELEMS;
    v16bf bf[4];
    #pragma unroll
    for (int c = 0; c < 4; ++c)                  // B fragments loaded once, reused 4x
      bf[c] = load_frag(WTc + ((wave << 6) + (c << 4) + m) * WTP + hk);
    #pragma unroll
    for (int r = 0; r < 4; ++r) {
      const v16bf a = load_frag(sA + ((r << 4) + m) * AP + ch * 32 + hk);
      #pragma unroll
      for (int c = 0; c < 4; ++c)
        acc[r][c] = __builtin_amdgcn_wmma_f32_16x16x32_bf16(
            false, a, false, bf[c], (short)0, acc[r][c], false, false);
    }
    __syncthreads();                             // buf[cur] consumed; safe to refill next iter
  }
  // epilogue: leaky-ReLU, write bf16 activations back into sA
  const int rbase = (lane >> 4) << 3;
  #pragma unroll
  for (int r = 0; r < 4; ++r) {
    #pragma unroll
    for (int c = 0; c < 4; ++c) {
      const int col = (wave << 6) + (c << 4) + m;
      #pragma unroll
      for (int v = 0; v < 8; ++v) {
        float xv = acc[r][c][v];
        xv = xv > 0.f ? xv : 0.01f * xv;
        sA[((r << 4) + rbase + v) * AP + col] = f2b(xv);
      }
    }
  }
  __syncthreads();
}

__global__ __launch_bounds__(NTHREADS)
void grandag_ws_kernel(const float* __restrict__ x,
                       const __bf16* __restrict__ ws,
                       const float* __restrict__ W2,
                       float* __restrict__ out) {
  __shared__ __align__(16) __bf16 sA [FT_N * AP];         // 66,560 B
  __shared__ __align__(16) __bf16 sWT[2 * CHUNK_ELEMS];   // 81,920 B (double buffer)
  const int tid  = threadIdx.x;
  const int wave = tid >> 5;
  const int lane = tid & 31;
  const int d    = blockIdx.y;
  const int n0   = blockIdx.x * FT_N;

  #pragma unroll
  for (int i = 0; i < (FT_N * Dn) / NTHREADS; ++i) {      // mask layer
    int idx = tid + i * NTHREADS;
    int row = idx >> 7, col = idx & 127;
    float v = x[(long)(n0 + row) * Dn + col];
    if (col == d) v = 0.f;
    sA[row * AP + col] = f2b(v);
  }
  __syncthreads();
  lc_gemm_ws<Dn>(ws + ((long)d * 4) * CHUNK_ELEMS,                 sA, sWT, tid, wave, lane);
  lc_gemm_ws<Hn>(ws + ((long)Dn * 4 + (long)d * 16) * CHUNK_ELEMS, sA, sWT, tid, wave, lane);

  // layer 2: [64x512] . W2[d] -> out[n0+row, d]
  float* red = (float*)sWT;
  const float* w2 = W2 + (long)d * Hn;
  {
    const int row = tid >> 2, seg = tid & 3;
    const __bf16* ar = sA + row * AP + seg * 128;
    const float*  wr = w2 + seg * 128;
    float s = 0.f;
    #pragma unroll 8
    for (int i = 0; i < 128; ++i) s += (float)ar[i] * wr[i];
    red[tid] = s;
  }
  __syncthreads();
  if (tid < FT_N) {
    float s = red[tid * 4] + red[tid * 4 + 1] + red[tid * 4 + 2] + red[tid * 4 + 3];
    out[(long)(n0 + tid) * Dn + d] = s;
  }
}

// ---------------- fallback (ws too small): inline convert ----------------
template<int K>
static __device__ __forceinline__ void lc_gemm(const float* __restrict__ Wg,
                                               __bf16* sA, __bf16* sWT,
                                               int tid, int wave, int lane) {
  v8f acc[2][4] = {};
  const int m  = lane & 15;
  const int hk = (lane >> 4) << 3;
  #pragma unroll 1
  for (int ch = 0; ch < K / 32; ++ch) {
    const int c0 = ch << 5;
    __syncthreads();
    #pragma unroll
    for (int i = 0; i < 16; ++i) {
      const int idx = tid + i * NTHREADS;
      const int k   = idx >> 7;
      const int n4  = (idx & 127) << 2;
      const float4 v = *(const float4*)(Wg + (long)(c0 + k) * Hn + n4);
      sWT[(n4 + 0) * WTP + k] = f2b(v.x);
      sWT[(n4 + 1) * WTP + k] = f2b(v.y);
      sWT[(n4 + 2) * WTP + k] = f2b(v.z);
      sWT[(n4 + 3) * WTP + k] = f2b(v.w);
      if (ch + 1 < K / 32)
        __builtin_prefetch(Wg + (long)(c0 + 32 + k) * Hn + n4, 0, 3);
    }
    __syncthreads();
    const v16bf a0 = load_frag(sA + (0 * 16 + m) * AP + c0 + hk);
    const v16bf a1 = load_frag(sA + (1 * 16 + m) * AP + c0 + hk);
    #pragma unroll
    for (int c = 0; c < 4; ++c) {
      const int n = (wave << 6) + (c << 4) + m;
      const v16bf b = load_frag(sWT + n * WTP + hk);
      acc[0][c] = __builtin_amdgcn_wmma_f32_16x16x32_bf16(
          false, a0, false, b, (short)0, acc[0][c], false, false);
      acc[1][c] = __builtin_amdgcn_wmma_f32_16x16x32_bf16(
          false, a1, false, b, (short)0, acc[1][c], false, false);
    }
  }
  __syncthreads();
  const int rbase = (lane >> 4) << 3;
  #pragma unroll
  for (int r = 0; r < 2; ++r)
    #pragma unroll
    for (int c = 0; c < 4; ++c) {
      const int col = (wave << 6) + (c << 4) + m;
      #pragma unroll
      for (int v = 0; v < 8; ++v) {
        float xv = acc[r][c][v];
        xv = xv > 0.f ? xv : 0.01f * xv;
        sA[((r << 4) + rbase + v) * AP + col] = f2b(xv);
      }
    }
  __syncthreads();
}

__global__ __launch_bounds__(NTHREADS)
void grandag_fused_kernel(const float* __restrict__ x,
                          const float* __restrict__ W0,
                          const float* __restrict__ W1,
                          const float* __restrict__ W2,
                          float* __restrict__ out) {
  __shared__ __align__(16) __bf16 sA [FB_N * AP];
  __shared__ __align__(16) __bf16 sWT[Hn * WTP];
  const int tid  = threadIdx.x;
  const int wave = tid >> 5;
  const int lane = tid & 31;
  const int d    = blockIdx.y;
  const int n0   = blockIdx.x * FB_N;
  #pragma unroll
  for (int i = 0; i < (FB_N * Dn) / NTHREADS; ++i) {
    int idx = tid + i * NTHREADS;
    int row = idx >> 7, col = idx & 127;
    float v = x[(long)(n0 + row) * Dn + col];
    if (col == d) v = 0.f;
    sA[row * AP + col] = f2b(v);
  }
  lc_gemm<Dn>(W0 + (long)d * Dn * Hn, sA, sWT, tid, wave, lane);
  lc_gemm<Hn>(W1 + (long)d * Hn * Hn, sA, sWT, tid, wave, lane);
  float* red = (float*)sWT;
  const float* w2 = W2 + (long)d * Hn;
  {
    const int row = tid >> 3, seg = tid & 7;
    const __bf16* ar = sA + row * AP + seg * 64;
    const float*  wr = w2 + seg * 64;
    float s = 0.f;
    #pragma unroll 8
    for (int i = 0; i < 64; ++i) s += (float)ar[i] * wr[i];
    red[tid] = s;
  }
  __syncthreads();
  if (tid < FB_N) {
    float s = 0.f;
    #pragma unroll
    for (int j = 0; j < 8; ++j) s += red[tid * 8 + j];
    out[(long)(n0 + tid) * Dn + d] = s;
  }
}

extern "C" void kernel_launch(void* const* d_in, const int* in_sizes, int n_in,
                              void* d_out, int out_size, void* d_ws, size_t ws_size,
                              hipStream_t stream) {
  (void)in_sizes; (void)n_in; (void)out_size;
  const float* x  = (const float*)d_in[0];
  const float* W0 = (const float*)d_in[1];
  const float* W1 = (const float*)d_in[2];
  const float* W2 = (const float*)d_in[3];
  float* out = (float*)d_out;
  dim3 block(NTHREADS);
  const size_t WS_NEED = (size_t)(4 + 16) * Dn * CHUNK_BYTES;   // ~100 MB
  if (ws_size >= WS_NEED && d_ws != nullptr) {
    dim3 cgrid(20, Dn);
    convert_weights_kernel<<<cgrid, block, 0, stream>>>(W0, W1, (__bf16*)d_ws);
    dim3 grid(Nn / FT_N, Dn);
    grandag_ws_kernel<<<grid, block, 0, stream>>>(x, (const __bf16*)d_ws, W2, out);
  } else {
    dim3 grid(Nn / FB_N, Dn);
    grandag_fused_kernel<<<grid, block, 0, stream>>>(x, W0, W1, W2, out);
  }
}
